// AttentionLayerSAP_QK_14860586844587
// MI455X (gfx1250) — matrix operand order repaired and checked by reading, compiled, not run on hardware
//
#include <hip/hip_runtime.h>
#include <hip/hip_bf16.h>

typedef __attribute__((ext_vector_type(16))) _Float16 v16h;
typedef __attribute__((ext_vector_type(8)))  _Float16 v8h;
typedef __attribute__((ext_vector_type(8)))  float    v8f;
typedef int async_b128_t __attribute__((vector_size(16)));  // builtin's pointee type

#define NUM_HEADS 12
#define DH        64
#define D_MODEL   768
#define SEQ       2048
#define BATCH     2
#define NEG_INF_VAL (-10000.0f)

#if defined(__has_builtin)
#if __has_builtin(__builtin_amdgcn_global_load_async_to_lds_b128)
#define HAVE_ASYNC_LDS 1
#endif
#endif
#ifndef HAVE_ASYNC_LDS
#define HAVE_ASYNC_LDS 0
#endif

static __device__ __forceinline__ v8f wmma_f16(v16h a, v16h b, v8f c) {
  // D = A(16x32 f16) * B(32x16 f16) + C(16x16 f32)
  return __builtin_amdgcn_wmma_f32_16x16x32_f16(
      /*neg_a=*/false, a, /*neg_b=*/false, b,
      /*c_mod=*/(short)0, c, /*reuse_a=*/false, /*reuse_b=*/false);
}

// ---------------------------------------------------------------------------
// Kernel 1: C = X[4096x768] * W[768x768] + bias, output f16 in [B][H][S][dh].
// Block = 256 threads (8 waves), tile 128x128, K-step 32.
// X tile staged f32 via async global->LDS copy (ASYNCcnt) when available;
// W tile staged sync (needs transpose scatter) with f32->f16 conversion.
// ---------------------------------------------------------------------------
__global__ __launch_bounds__(256) void proj_qk_kernel(
    const float* __restrict__ X, const float* __restrict__ W,
    const float* __restrict__ bias, _Float16* __restrict__ outH) {
  __shared__ __align__(16) float    xsf[128][36];  // [row][k] f32, 144B stride
  __shared__ __align__(16) _Float16 wt[128][40];   // [col][k] f16 transposed

  const int tid  = threadIdx.x;
  const int lane = tid & 31;
  const int w    = tid >> 5;
  const int hi   = lane >> 4;   // half-wave select
  const int ln   = lane & 15;
  const int mBase = blockIdx.y * 128;
  const int nBase = blockIdx.x * 128;
  const int waveM = (w & 3) * 32;
  const int waveN = (w >> 2) * 64;

  v8f acc[2][4];
#pragma unroll
  for (int i = 0; i < 2; ++i)
#pragma unroll
    for (int j = 0; j < 4; ++j) acc[i][j] = v8f{};

  for (int kb = 0; kb < D_MODEL; kb += 32) {
    // ---- Stage X tile 128x32 f32 into LDS (async if available) ----
#pragma unroll
    for (int it = 0; it < 4; ++it) {
      const int pos = tid + it * 256;
      const int row = pos >> 3;            // 8 float4 per row
      const int c4  = (pos & 7) << 2;
      const float* gsrc = X + (size_t)(mBase + row) * D_MODEL + kb + c4;
#if HAVE_ASYNC_LDS
      __builtin_amdgcn_global_load_async_to_lds_b128(
          (__attribute__((address_space(1))) async_b128_t*)gsrc,
          (__attribute__((address_space(3))) async_b128_t*)&xsf[row][c4],
          /*offset=*/0, /*cpol=*/0);
#else
      *(float4*)&xsf[row][c4] = *(const float4*)gsrc;
#endif
    }
    // ---- Stage W tile 32x128 transposed, f32 -> f16 ----
#pragma unroll
    for (int it = 0; it < 4; ++it) {
      const int pos = tid + it * 256;
      const int kk  = pos >> 5;            // 32 float4 per row
      const int c4  = (pos & 31) << 2;
      const float4 v = *(const float4*)(W + (size_t)(kb + kk) * D_MODEL + nBase + c4);
      wt[c4 + 0][kk] = (_Float16)v.x;
      wt[c4 + 1][kk] = (_Float16)v.y;
      wt[c4 + 2][kk] = (_Float16)v.z;
      wt[c4 + 3][kk] = (_Float16)v.w;
    }
#if HAVE_ASYNC_LDS
#if __has_builtin(__builtin_amdgcn_s_wait_asynccnt)
    __builtin_amdgcn_s_wait_asynccnt(0);
#else
    asm volatile("s_wait_asynccnt 0" ::: "memory");
#endif
#endif
    __syncthreads();

    // A fragments: elements 0..7 = K hi*8..hi*8+7, elements 8..15 = K 16+hi*8..
    v16h a[2], bfrag[4];
#pragma unroll
    for (int i = 0; i < 2; ++i) {
      const int row = waveM + i * 16 + ln;
      const float4 f0 = *(const float4*)&xsf[row][hi * 8];
      const float4 f1 = *(const float4*)&xsf[row][hi * 8 + 4];
      const float4 f2 = *(const float4*)&xsf[row][16 + hi * 8];
      const float4 f3 = *(const float4*)&xsf[row][16 + hi * 8 + 4];
      a[i][0]  = (_Float16)f0.x; a[i][1]  = (_Float16)f0.y;
      a[i][2]  = (_Float16)f0.z; a[i][3]  = (_Float16)f0.w;
      a[i][4]  = (_Float16)f1.x; a[i][5]  = (_Float16)f1.y;
      a[i][6]  = (_Float16)f1.z; a[i][7]  = (_Float16)f1.w;
      a[i][8]  = (_Float16)f2.x; a[i][9]  = (_Float16)f2.y;
      a[i][10] = (_Float16)f2.z; a[i][11] = (_Float16)f2.w;
      a[i][12] = (_Float16)f3.x; a[i][13] = (_Float16)f3.y;
      a[i][14] = (_Float16)f3.z; a[i][15] = (_Float16)f3.w;
    }
    // B fragments: per lane column n = ln, elements 0..7 = K hi*8.., 8..15 = K 16+hi*8..
#pragma unroll
    for (int j = 0; j < 4; ++j) {
      const int col = waveN + j * 16 + ln;
      const v8h lo = *reinterpret_cast<const v8h*>(&wt[col][hi * 8]);
      const v8h up = *reinterpret_cast<const v8h*>(&wt[col][16 + hi * 8]);
#pragma unroll
      for (int e = 0; e < 8; ++e) { bfrag[j][e] = lo[e]; bfrag[j][8 + e] = up[e]; }
    }
#pragma unroll
    for (int i = 0; i < 2; ++i)
#pragma unroll
      for (int j = 0; j < 4; ++j)
        acc[i][j] = wmma_f16(a[i], bfrag[j], acc[i][j]);
    __syncthreads();
  }

  // C layout: lane col n = ln; VGPR r -> row r + 8*hi. Add bias, store f16
  // into head-major [B][H][S][dh].
#pragma unroll
  for (int i = 0; i < 2; ++i) {
#pragma unroll
    for (int j = 0; j < 4; ++j) {
      const int col = nBase + waveN + j * 16 + ln;
      const float bv = bias[col];
      const int hh = col >> 6, dj = col & 63;
#pragma unroll
      for (int r = 0; r < 8; ++r) {
        const int rowG = mBase + waveM + i * 16 + r + 8 * hi;
        const int bb = rowG >> 11;       // / SEQ
        const int ss = rowG & (SEQ - 1);
        outH[(((size_t)bb * NUM_HEADS + hh) * SEQ + ss) * DH + dj] =
            (_Float16)(acc[i][j][r] + bv);
      }
    }
  }
}

// ---------------------------------------------------------------------------
// Kernel 2: scores = (Q*K^T)/8 + mask-bias, row softmax, write probs (f32).
// Grid (S/16, H, B); block 256 = 8 waves; each wave: 16 score tiles (16x256),
// full 16x256 slice held in 128 accumulator VGPRs.
// ---------------------------------------------------------------------------
__global__ __launch_bounds__(256) void attn_softmax_kernel(
    const _Float16* __restrict__ Q, const _Float16* __restrict__ K,
    const int* __restrict__ mask, float* __restrict__ out) {
  __shared__ float red[8][16];  // [wave][row] cross-wave reduction

  const int tid  = threadIdx.x;
  const int lane = tid & 31;
  const int w    = tid >> 5;
  const int hi   = lane >> 4;
  const int ln   = lane & 15;
  const int qBase = blockIdx.x * 16;
  const int h = blockIdx.y;
  const int b = blockIdx.z;
  const int kWaveBase = w * 256;
  const size_t headBase = ((size_t)b * NUM_HEADS + h) * SEQ;

  // Prefetch the (L2-resident) mask rows this wave will consume, so the
  // fetches overlap the score WMMAs (lowers to global_prefetch_b8).
#pragma unroll
  for (int r = 0; r < 8; ++r) {
    const int row = qBase + r + 8 * hi;
    __builtin_prefetch(mask + ((size_t)b * SEQ + row) * SEQ + kWaveBase + ln, 0, 1);
  }

  // A fragments (q rows, K = dh = 64 -> two 16x32 fragments), direct global.
  const _Float16* qp = Q + (headBase + qBase + ln) * DH;
  v16h a0, a1;
  {
    const v8h l0 = *(const v8h*)(qp + hi * 8);
    const v8h u0 = *(const v8h*)(qp + 16 + hi * 8);
    const v8h l1 = *(const v8h*)(qp + 32 + hi * 8);
    const v8h u1 = *(const v8h*)(qp + 48 + hi * 8);
#pragma unroll
    for (int e = 0; e < 8; ++e) {
      a0[e] = l0[e]; a0[8 + e] = u0[e];
      a1[e] = l1[e]; a1[8 + e] = u1[e];
    }
  }

  v8f acc[16];
#pragma unroll
  for (int t = 0; t < 16; ++t) acc[t] = v8f{};

  // Score GEMM: B fragment lane = key column, K (=d) contiguous in k layout.
#pragma unroll
  for (int t = 0; t < 16; ++t) {
    const _Float16* kp = K + (headBase + kWaveBase + t * 16 + ln) * DH;
    const v8h l0 = *(const v8h*)(kp + hi * 8);
    const v8h u0 = *(const v8h*)(kp + 16 + hi * 8);
    const v8h l1 = *(const v8h*)(kp + 32 + hi * 8);
    const v8h u1 = *(const v8h*)(kp + 48 + hi * 8);
    v16h b0, b1;
#pragma unroll
    for (int e = 0; e < 8; ++e) {
      b0[e] = l0[e]; b0[8 + e] = u0[e];
      b1[e] = l1[e]; b1[8 + e] = u1[e];
    }
    acc[t] = wmma_f16(a0, b0, acc[t]);
    acc[t] = wmma_f16(a1, b1, acc[t]);
  }

  // Scale + additive mask + per-lane row max (lane owns rows r+8*hi, col ln).
  float pm[8];
#pragma unroll
  for (int r = 0; r < 8; ++r) pm[r] = -3.0e38f;
#pragma unroll
  for (int r = 0; r < 8; ++r) {
    const int row = qBase + r + 8 * hi;
    const int* mrow = mask + ((size_t)b * SEQ + row) * SEQ + kWaveBase + ln;
#pragma unroll
    for (int t = 0; t < 16; ++t) {
      const int mv = mrow[t * 16];
      const float v = acc[t][r] * 0.125f + (1.0f - (float)mv) * NEG_INF_VAL;
      acc[t][r] = v;
      pm[r] = fmaxf(pm[r], v);
    }
  }
  // Reduce across the 16 lanes of each half-wave (columns of this tile strip).
#pragma unroll
  for (int r = 0; r < 8; ++r) {
#pragma unroll
    for (int off = 1; off < 16; off <<= 1)
      pm[r] = fmaxf(pm[r], __shfl_xor(pm[r], off, 32));
  }
  if (ln == 0) {
#pragma unroll
    for (int r = 0; r < 8; ++r) red[w][8 * hi + r] = pm[r];
  }
  __syncthreads();
  float rowmax[8];
#pragma unroll
  for (int r = 0; r < 8; ++r) {
    float m = red[0][r + 8 * hi];
#pragma unroll
    for (int ww = 1; ww < 8; ++ww) m = fmaxf(m, red[ww][r + 8 * hi]);
    rowmax[r] = m;
  }
  __syncthreads();

  // exp + row sum
  float ps[8];
#pragma unroll
  for (int r = 0; r < 8; ++r) {
    float s = 0.0f;
#pragma unroll
    for (int t = 0; t < 16; ++t) {
      const float e = __expf(acc[t][r] - rowmax[r]);
      acc[t][r] = e;
      s += e;
    }
    ps[r] = s;
  }
#pragma unroll
  for (int r = 0; r < 8; ++r) {
#pragma unroll
    for (int off = 1; off < 16; off <<= 1)
      ps[r] += __shfl_xor(ps[r], off, 32);
  }
  if (ln == 0) {
#pragma unroll
    for (int r = 0; r < 8; ++r) red[w][8 * hi + r] = ps[r];
  }
  __syncthreads();

  // Normalize and write probs[b][h][row][col] (f32, written exactly once).
#pragma unroll
  for (int r = 0; r < 8; ++r) {
    float s = 0.0f;
#pragma unroll
    for (int ww = 0; ww < 8; ++ww) s += red[ww][r + 8 * hi];
    const float inv = 1.0f / s;
    float* orow = out + (headBase + qBase + r + 8 * hi) * (size_t)SEQ + kWaveBase + ln;
#pragma unroll
    for (int t = 0; t < 16; ++t) orow[t * 16] = acc[t][r] * inv;
  }
}

// ---------------------------------------------------------------------------
extern "C" void kernel_launch(void* const* d_in, const int* in_sizes, int n_in,
                              void* d_out, int out_size, void* d_ws, size_t ws_size,
                              hipStream_t stream) {
  const float* x    = (const float*)d_in[0];
  const int*   mask = (const int*)d_in[1];
  const float* Wq   = (const float*)d_in[2];
  const float* bq   = (const float*)d_in[3];
  const float* Wk   = (const float*)d_in[4];
  const float* bk   = (const float*)d_in[5];
  float* out = (float*)d_out;

  _Float16* qh = (_Float16*)d_ws;                                   // 6.29 MB
  _Float16* kh = qh + (size_t)BATCH * NUM_HEADS * SEQ * DH;         // 6.29 MB

  dim3 g1(D_MODEL / 128, (BATCH * SEQ) / 128);  // (6, 32)
  dim3 b1(256);
  proj_qk_kernel<<<g1, b1, 0, stream>>>(x, Wq, bq, qh);
  proj_qk_kernel<<<g1, b1, 0, stream>>>(x, Wk, bk, kh);

  dim3 g2(SEQ / 16, NUM_HEADS, BATCH);          // (128, 12, 2)
  dim3 b2(256);
  attn_softmax_kernel<<<g2, b2, 0, stream>>>(qh, kh, mask, out);
}
